// TransDiffuserIntegrated_78288663872334
// MI455X (gfx1250) — matrix operand order repaired and checked by reading, not compiled, not run on hardware
//
#include <hip/hip_runtime.h>
#include <math.h>

// ---------------------------------------------------------------------------
// CDNA5 (gfx1250, wave32) WMMA types & helpers
// ---------------------------------------------------------------------------
typedef __attribute__((ext_vector_type(16))) _Float16 v16h;
typedef __attribute__((ext_vector_type(4)))  _Float16 v4h;
typedef __attribute__((ext_vector_type(8)))  float    v8f;
typedef __attribute__((ext_vector_type(4)))  unsigned int u32x4;
typedef __attribute__((ext_vector_type(8)))  int      i32x8;
typedef __attribute__((ext_vector_type(4)))  int      i32x4;

__device__ __forceinline__ v8f wmma32(v16h a, v16h b, v8f c) {
  // D = A(16x32,f16) * B(32x16,f16) + C(16x16,f32)
  return __builtin_amdgcn_wmma_f32_16x16x32_f16(false, a, false, b, (short)0, c,
                                                false, false);
}

__device__ __forceinline__ float geluf(float x) {
  // jax.nn.gelu default (approximate=True, tanh form)
  float x3 = x * x * x;
  return 0.5f * x * (1.0f + tanhf(0.7978845608028654f * (x + 0.044715f * x3)));
}
__device__ __forceinline__ float siluf(float x) { return x / (1.0f + expf(-x)); }

// Model constants
#define DM    768
#define NHD   12
#define HD    64
#define BAT   8
#define NTOK  544        // 256 traj + 64 A + 32 B + 192 C
#define FFH   3072

// ---------------------------------------------------------------------------
// Small elementwise kernels
// ---------------------------------------------------------------------------
__global__ void k_temb(const int* __restrict__ t, float* __restrict__ temb) {
  int b = blockIdx.x, i = threadIdx.x;
  float tv = (float)t[b];
  if (i < 128) {
    float f = expf(-9.210340371976184f * (float)i / 128.0f);
    temb[b * 256 + i] = cosf(tv * f);
  } else {
    int j = i - 128;
    float f = expf(-9.210340371976184f * (float)j / 128.0f);
    temb[b * 256 + i] = sinf(tv * f);
  }
}

__global__ void k_silu(const float* __restrict__ x, float* __restrict__ y, int n) {
  int i = blockIdx.x * 256 + threadIdx.x;
  if (i < n) y[i] = siluf(x[i]);
}

__global__ void k_copy(const float* __restrict__ s, float* __restrict__ d, int n) {
  for (int i = blockIdx.x * 256 + threadIdx.x; i < n; i += gridDim.x * 256) d[i] = s[i];
}

__global__ void k_zero_f(float* __restrict__ p, int n) {
  int i = blockIdx.x * 256 + threadIdx.x;
  if (i < n) p[i] = 0.0f;
}

__global__ void k_zero_i(int* __restrict__ p, int n) {
  int i = blockIdx.x * 32 + threadIdx.x;
  if (i < n) p[i] = 0;
}

// LayerNorm (+ optional adaLN modulate): y = ln(x)*(1+scale[b,:]) + shift[b,:]
__global__ void k_ln_mod(const float* __restrict__ X, float* __restrict__ Y,
                         const float* __restrict__ shift, const float* __restrict__ scale,
                         int bStride, int rowsPerBatch, int M) {
  int row = blockIdx.x;
  if (row >= M) return;
  const float* x = X + (long)row * DM;
  __shared__ float red[256];
  int tid = threadIdx.x;
  float v0 = x[tid], v1 = x[tid + 256], v2 = x[tid + 512];
  red[tid] = v0 + v1 + v2; __syncthreads();
  for (int d = 128; d; d >>= 1) { if (tid < d) red[tid] += red[tid + d]; __syncthreads(); }
  float mu = red[0] * (1.0f / 768.0f); __syncthreads();
  float d0 = v0 - mu, d1 = v1 - mu, d2 = v2 - mu;
  red[tid] = d0 * d0 + d1 * d1 + d2 * d2; __syncthreads();
  for (int d = 128; d; d >>= 1) { if (tid < d) red[tid] += red[tid + d]; __syncthreads(); }
  float inv = rsqrtf(red[0] * (1.0f / 768.0f) + 1e-6f);
  float n0 = d0 * inv, n1 = d1 * inv, n2 = d2 * inv;
  float* y = Y + (long)row * DM;
  if (scale) {
    int b = row / rowsPerBatch;
    const float* sc = scale + (long)b * bStride;
    const float* sh = shift + (long)b * bStride;
    y[tid]       = n0 * (1.0f + sc[tid])       + sh[tid];
    y[tid + 256] = n1 * (1.0f + sc[tid + 256]) + sh[tid + 256];
    y[tid + 512] = n2 * (1.0f + sc[tid + 512]) + sh[tid + 512];
  } else {
    y[tid] = n0; y[tid + 256] = n1; y[tid + 512] = n2;
  }
}

// Assemble X = [traj(+ftime) | A | B | C]
__global__ void k_assemble(const float* __restrict__ traj, const float* __restrict__ ftime,
                           const float* __restrict__ GA, const float* __restrict__ GB,
                           const float* __restrict__ GC, float* __restrict__ X) {
  int row = blockIdx.x;
  int b = row / NTOK, tk = row % NTOK;
  float* dst = X + (long)row * DM;
  int tid = threadIdx.x;
  #pragma unroll
  for (int j = 0; j < 3; ++j) {
    int c = tid + j * 256;
    float v;
    if (tk < 256)      v = traj[((long)(b * 256 + tk)) * DM + c] + ftime[(tk & 7) * DM + c];
    else if (tk < 320) v = GA[((long)(b * 64  + (tk - 256))) * DM + c];
    else if (tk < 352) v = GB[((long)(b * 32  + (tk - 320))) * DM + c];
    else               v = GC[((long)(b * 192 + (tk - 352))) * DM + c];
    dst[c] = v;
  }
}

__global__ void k_outcopy(const float* __restrict__ heado, float* __restrict__ out) {
  int i = blockIdx.x * 256 + threadIdx.x;
  if (i >= BAT * 256 * 14) return;
  int b = i / (256 * 14);
  int r = (i / 14) % 256;
  int j = i % 14;
  out[i] = heado[((long)(b * NTOK + r)) * 14 + j];
}

// ---------------------------------------------------------------------------
// Generic WMMA GEMM: C = epi(A[M,K] @ W[K,N]); epi = +bias, act, *gate[b,n], +resid
// Requires M%16==0 (host pads conditioning GEMMs to 16 rows) -> NO row guards.
// Uniform fast path for K%64==0 && N%128==0: zero guards, float4-vectorized
// A staging, unguarded coalesced B loads, global_prefetch on next panel.
// Slow path (K=7 traj / N=14 head only): clamped address + value select.
// ---------------------------------------------------------------------------
__global__ void k_gemm(const float* __restrict__ A, const float* __restrict__ W,
                       const float* __restrict__ bias, const float* __restrict__ gate,
                       int gateStride, int rowsPerBatch, const float* __restrict__ resid,
                       float* __restrict__ C, int M, int K, int N, int act) {
  __shared__ _Float16 As[16 * 72];
  int tid = threadIdx.x, wave = tid >> 5, lane = tid & 31;
  int rowBase = blockIdx.y * 16;
  int colBase = blockIdx.x * 128 + wave * 16;
  int lr = lane & 15;
  int khalf = (lane & 16) ? 8 : 0;
  int kh16  = (lane & 16) ? 16 : 0;
  int n = colBase + lr;
  v8f acc = {};
  const bool fast = ((K & 63) == 0) && ((N & 127) == 0);   // uniform branch
  if (fast) {
    int sr = tid >> 4;            // staging row 0..15
    int sc = (tid & 15) * 4;      // staging col group (float4)
    const float* arow = A + (long)(rowBase + sr) * K;
    for (int k0 = 0; k0 < K; k0 += 64) {
      float4 f = *(const float4*)(arow + k0 + sc);
      v4h h4;
      h4[0] = (_Float16)f.x; h4[1] = (_Float16)f.y;
      h4[2] = (_Float16)f.z; h4[3] = (_Float16)f.w;
      *(v4h*)(As + sr * 72 + sc) = h4;
      __syncthreads();
      #pragma unroll
      for (int kk = 0; kk < 64; kk += 32) {
        v16h a;
        const _Float16* ar = As + lr * 72 + kk + khalf;
        #pragma unroll
        for (int t = 0; t < 16; ++t) a[t] = ar[(t & 7) + ((t & 8) ? 16 : 0)];
        v16h bf;
        int kb = k0 + kk + kh16;
        if (kb + 64 < K) __builtin_prefetch(&W[(long)(kb + 64) * N + n], 0, 0);
        #pragma unroll
        for (int t = 0; t < 16; ++t) bf[t] = (_Float16)W[(long)(k0 + kk + khalf + (t & 7) + ((t & 8) ? 16 : 0)) * N + n];
        acc = wmma32(a, bf, acc);
      }
      __syncthreads();
    }
    #pragma unroll
    for (int vv = 0; vv < 8; ++vv) {
      int m = rowBase + vv + ((lane & 16) ? 8 : 0);
      float x = acc[vv];
      if (bias) x += bias[n];
      if (act == 1) x = geluf(x);
      else if (act == 2) x = siluf(x);
      if (gate) x *= gate[(long)(m / rowsPerBatch) * gateStride + n];
      if (resid) x += resid[(long)m * N + n];
      C[(long)m * N + n] = x;
    }
  } else {
    int nc = n < N ? n : N - 1;
    bool cv = n < N;
    for (int k0 = 0; k0 < K; k0 += 64) {
      #pragma unroll
      for (int i = 0; i < 4; ++i) {
        int idx = tid + i * 256;
        int r = idx >> 6, kk = idx & 63;
        int gk = k0 + kk;
        int gkc = gk < K ? gk : K - 1;
        float xv = A[(long)(rowBase + r) * K + gkc];
        As[r * 72 + kk] = (_Float16)((gk < K) ? xv : 0.0f);
      }
      __syncthreads();
      #pragma unroll
      for (int kk = 0; kk < 64; kk += 32) {
        v16h a;
        const _Float16* ar = As + lr * 72 + kk + khalf;
        #pragma unroll
        for (int t = 0; t < 16; ++t) a[t] = ar[(t & 7) + ((t & 8) ? 16 : 0)];
        v16h bf;
        int kb = k0 + kk + kh16;
        #pragma unroll
        for (int t = 0; t < 16; ++t) {
          int gk = k0 + kk + khalf + (t & 7) + ((t & 8) ? 16 : 0);
          int gkc = gk < K ? gk : K - 1;
          float wv = W[(long)gkc * N + nc];
          bf[t] = (_Float16)((cv && gk < K) ? wv : 0.0f);
        }
        acc = wmma32(a, bf, acc);
      }
      __syncthreads();
    }
    if (cv) {
      #pragma unroll
      for (int vv = 0; vv < 8; ++vv) {
        int m = rowBase + vv + ((lane & 16) ? 8 : 0);
        float x = acc[vv];
        if (bias) x += bias[n];
        if (act == 1) x = geluf(x);
        else if (act == 2) x = siluf(x);
        if (gate) x *= gate[(long)(m / rowsPerBatch) * gateStride + n];
        if (resid) x += resid[(long)m * N + n];
        C[(long)m * N + n] = x;
      }
    }
  }
}

// ---------------------------------------------------------------------------
// Flash attention (online softmax) with WMMA. One (batch,head) per blockIdx.x,
// one 16-row q tile per wave. P relayout C->A via per-wave LDS + s_wait_dscnt.
// Requires Nq%16==0, Nk%32==0 (holds: 32/192/544 q, 64/192/544 k).
// ---------------------------------------------------------------------------
__global__ void k_attn(const float* __restrict__ qB, int qStride,
                       const float* __restrict__ kB, int kStride,
                       const float* __restrict__ vB, int vStride,
                       float* __restrict__ oB, int oStride, int Nq, int Nk) {
  __shared__ _Float16 Pb[8][16 * 40];
  int bh = blockIdx.x, b = bh / NHD, h = bh % NHD;
  int wave = threadIdx.x >> 5, lane = threadIdx.x & 31;
  int qt = blockIdx.y * 8 + wave;
  if (qt * 16 >= Nq) return;
  const float* q = qB + (long)b * Nq * qStride + h * HD;
  const float* kM = kB + (long)b * Nk * kStride + h * HD;
  const float* vM = vB + (long)b * Nk * vStride + h * HD;
  float* o = oB + (long)b * Nq * oStride + h * HD;
  _Float16* P = &Pb[wave][0];
  int lr = lane & 15;
  int khalf = (lane & 16) ? 8 : 0;
  int kh16  = (lane & 16) ? 16 : 0;
  // Q fragments (A layout), channels 0..31 and 32..63
  const float* qr = q + (long)(qt * 16 + lr) * qStride;
  v16h aq0, aq1;
  #pragma unroll
  for (int t = 0; t < 16; ++t) {
    int c = kh16 + t;
    aq0[t] = (_Float16)qr[c];
    aq1[t] = (_Float16)qr[32 + c];
  }
  v8f z = {};
  v8f oAcc[4]; oAcc[0] = z; oAcc[1] = z; oAcc[2] = z; oAcc[3] = z;
  float msf[8], lsf[8];
  #pragma unroll
  for (int i = 0; i < 8; ++i) { msf[i] = -1e30f; lsf[i] = 0.0f; }

  for (int kb = 0; kb < Nk; kb += 32) {
    // S = Q @ K^T for 32 keys (two 16-key subtiles)
    v8f s0 = {}, s1 = {};
    {
      const float* kr = kM + (long)(kb + lr) * kStride;
      v16h b0, b1;
      #pragma unroll
      for (int t = 0; t < 16; ++t) { b0[t] = (_Float16)kr[kh16 + t]; b1[t] = (_Float16)kr[32 + kh16 + t]; }
      s0 = wmma32(aq0, b0, s0); s0 = wmma32(aq1, b1, s0);
    }
    {
      const float* kr = kM + (long)(kb + 16 + lr) * kStride;
      v16h b0, b1;
      #pragma unroll
      for (int t = 0; t < 16; ++t) { b0[t] = (_Float16)kr[kh16 + t]; b1[t] = (_Float16)kr[32 + kh16 + t]; }
      s1 = wmma32(aq0, b0, s1); s1 = wmma32(aq1, b1, s1);
    }
    // online softmax over this 16x32 block
    #pragma unroll
    for (int vv = 0; vv < 8; ++vv) {
      float a0 = s0[vv] * 0.125f, a1 = s1[vv] * 0.125f;   // HD^-0.5
      float mx = fmaxf(a0, a1);
      #pragma unroll
      for (int d = 1; d < 16; d <<= 1) mx = fmaxf(mx, __shfl_xor(mx, d, 32));
      float mn = fmaxf(msf[vv], mx);
      float p0 = expf(a0 - mn), p1 = expf(a1 - mn);
      float corr = expf(msf[vv] - mn);
      float rs = p0 + p1;
      #pragma unroll
      for (int d = 1; d < 16; d <<= 1) rs += __shfl_xor(rs, d, 32);
      lsf[vv] = lsf[vv] * corr + rs;
      msf[vv] = mn;
      oAcc[0][vv] *= corr; oAcc[1][vv] *= corr; oAcc[2][vv] *= corr; oAcc[3][vv] *= corr;
      int m = vv + ((lane & 16) ? 8 : 0);
      P[m * 40 + lr]      = (_Float16)p0;
      P[m * 40 + 16 + lr] = (_Float16)p1;
    }
    asm volatile("s_wait_dscnt 0" ::: "memory");   // per-wave LDS RAW fence
    // O += P(16x32) @ V(32x64)
    v16h ap;
    const _Float16* pr = P + lr * 40;
    #pragma unroll
    for (int t = 0; t < 16; ++t) ap[t] = pr[kh16 + t];
    #pragma unroll
    for (int ct = 0; ct < 4; ++ct) {
      v16h bv;
      int ch = ct * 16 + lr;
      #pragma unroll
      for (int t = 0; t < 16; ++t) bv[t] = (_Float16)vM[(long)(kb + kh16 + t) * vStride + ch];
      oAcc[ct] = wmma32(ap, bv, oAcc[ct]);
    }
  }
  #pragma unroll
  for (int ct = 0; ct < 4; ++ct)
    #pragma unroll
    for (int vv = 0; vv < 8; ++vv) {
      int m = qt * 16 + vv + ((lane & 16) ? 8 : 0);
      o[(long)m * oStride + ct * 16 + lr] = oAcc[ct][vv] / lsf[vv];
    }
}

// ---------------------------------------------------------------------------
// MoE routing: logits->softmax->argmax (top-1 switch), counts via atomics
// int ws layout: counts[0..3] cursors[4..7] off[8..12] total[13] tileExpert[16..127]
//                eidx[128..1663] srcRow[1792..3391]
// ---------------------------------------------------------------------------
#define IO_CNT 0
#define IO_CUR 4
#define IO_OFF 8
#define IO_TOT 13
#define IO_TIL 16
#define IO_EID 128
#define IO_SRC 1792

__global__ void k_router(const float* __restrict__ lnx, const float* __restrict__ Wr,
                         int R, int* __restrict__ ip, float* __restrict__ gval) {
  int wave = threadIdx.x >> 5, lane = threadIdx.x & 31;
  int row = blockIdx.x * 8 + wave;
  if (row >= R) return;
  const float* x = lnx + (long)row * DM;
  float a0 = 0, a1 = 0, a2 = 0, a3 = 0;
  for (int k = lane; k < DM; k += 32) {
    float xv = x[k]; const float* w = Wr + k * 4;
    a0 += xv * w[0]; a1 += xv * w[1]; a2 += xv * w[2]; a3 += xv * w[3];
  }
  #pragma unroll
  for (int d = 1; d < 32; d <<= 1) {
    a0 += __shfl_xor(a0, d, 32); a1 += __shfl_xor(a1, d, 32);
    a2 += __shfl_xor(a2, d, 32); a3 += __shfl_xor(a3, d, 32);
  }
  if (lane == 0) {
    float mx = fmaxf(fmaxf(a0, a1), fmaxf(a2, a3));
    float e0 = expf(a0 - mx), e1 = expf(a1 - mx), e2 = expf(a2 - mx), e3 = expf(a3 - mx);
    float s = e0 + e1 + e2 + e3;
    int best = 0; float bv = a0;
    if (a1 > bv) { bv = a1; best = 1; }
    if (a2 > bv) { bv = a2; best = 2; }
    if (a3 > bv) { bv = a3; best = 3; }
    float ge = (best == 0 ? e0 : best == 1 ? e1 : best == 2 ? e2 : e3) / s;
    ip[IO_EID + row] = best;
    gval[row] = ge;
    atomicAdd(&ip[IO_CNT + best], 1);
  }
}

__global__ void k_offsets(int* __restrict__ ip) {
  int off = 0, tt = 0;
  for (int e = 0; e < 4; ++e) {
    ip[IO_OFF + e] = off;
    int c = ip[IO_CNT + e];
    int tiles = (c + 15) >> 4;
    for (int t = 0; t < tiles; ++t) ip[IO_TIL + tt + t] = e;
    tt += tiles; off += tiles << 4;
    ip[IO_CUR + e] = 0;
  }
  ip[IO_OFF + 4] = off;
  ip[IO_TOT] = tt;
}

__global__ void k_init_src(int* __restrict__ ip, int n) {
  int i = blockIdx.x * 256 + threadIdx.x;
  if (i < n) ip[IO_SRC + i] = -1;
}

__global__ void k_scatter(int* __restrict__ ip, int R) {
  int i = blockIdx.x * 256 + threadIdx.x;
  if (i >= R) return;
  int e = ip[IO_EID + i];
  int pos = ip[IO_OFF + e] + atomicAdd(&ip[IO_CUR + e], 1);
  ip[IO_SRC + pos] = i;
}

__global__ void k_gather(const float* __restrict__ lnx, const int* __restrict__ ip,
                         _Float16* __restrict__ Xs, int maxRows) {
  int row = blockIdx.x;
  if (row >= maxRows) return;
  int src = ip[IO_SRC + row];
  int tid = threadIdx.x;
  _Float16* y = Xs + (long)row * DM;
  if (src < 0) {
    y[tid] = (_Float16)0.0f; y[tid + 256] = (_Float16)0.0f; y[tid + 512] = (_Float16)0.0f;
  } else {
    const float* x = lnx + (long)src * DM;
    y[tid] = (_Float16)x[tid]; y[tid + 256] = (_Float16)x[tid + 256]; y[tid + 512] = (_Float16)x[tid + 512];
  }
}

// ---------------------------------------------------------------------------
// Fused FFN: Out[...] += gate * (gelu(A @ W1 + b1) @ W2 + b2)
// Hidden 16x3072 f16 kept in LDS (never hits HBM); weights L2-resident.
// MoE mode: input tile staged via TDM tensor_load_to_lds with descriptor
// iteration (16 rows, global stride 768 halves -> LDS stride 776 halves for
// bank-conflict-free ds_load_b128 A fragments). DiT mode: manual f32->f16.
// ---------------------------------------------------------------------------
#define XA_LD 776
#define H_LD  3080
#define FFN_SMEM ((16 * XA_LD + 16 * H_LD) * 2)

__global__ void k_ffn(const float* __restrict__ Af32, const _Float16* __restrict__ Ah16,
                      const float* __restrict__ W1b, const float* __restrict__ b1,
                      const float* __restrict__ W2b, const float* __restrict__ b2,
                      float* __restrict__ Out,
                      const float* __restrict__ gateVec, int gateStride, int rowsPerBatch,
                      const float* __restrict__ gateRow, const int* __restrict__ ip,
                      long w1es, long w2es) {
  extern __shared__ char smem[];
  _Float16* Xa = (_Float16*)smem;
  _Float16* H  = Xa + 16 * XA_LD;
  int tile = blockIdx.x;
  bool moe = (ip != nullptr);
  if (moe && tile >= ip[IO_TOT]) return;
  int e = moe ? ip[IO_TIL + tile] : 0;
  const float* W1 = W1b + (long)e * w1es;
  const float* W2 = W2b + (long)e * w2es;
  int tid = threadIdx.x, wave = tid >> 5, lane = tid & 31, lr = lane & 15;
  int khalf = (lane & 16) ? 8 : 0;
  int kh16  = (lane & 16) ? 16 : 0;
  int hi8   = (lane & 16) ? 8 : 0;
  // ---- stage input tile into LDS (f16) ----
  if (moe) {
#if __has_builtin(__builtin_amdgcn_tensor_load_to_lds) && defined(__gfx1250__)
    // TDM: 16 iterations of a 1-row 768xf16 tile; LDS row stride padded to 776.
    if (wave == 0) {
      const _Float16* src = Ah16 + (long)tile * 16 * DM;
      unsigned long long ga = (unsigned long long)(const void*)src;
      unsigned int ldsOff = (unsigned int)(unsigned long long)(void*)Xa;
      u32x4 g0;
      g0[0] = 1u;                                        // count=1 user D#
      g0[1] = ldsOff;                                    // lds_addr
      g0[2] = (unsigned int)(ga & 0xffffffffu);          // global_addr lo
      g0[3] = (unsigned int)((ga >> 32) & 0x01ffffffu)   // global_addr hi (bits 56:32)
              | (2u << 30);                              // type=2 (image)
      i32x8 g1;
      g1[0] = (1 << 16) | (1 << 19);                     // data_size=2B, iterate_enable
      g1[1] = (int)(768u << 16);                         // tensor_dim0 = 768 (lo16)
      g1[2] = (int)(1u << 16);                           // dim0 hi=0 | tensor_dim1 = 1
      g1[3] = (int)(768u << 16);                         // dim1 hi=0 | tile_dim0 = 768
      g1[4] = 1;                                         // tile_dim1 = 1, tile_dim2 = 0
      g1[5] = 768;                                       // tensor_dim0_stride = 768
      g1[6] = 0;                                         // stride hi | dim1_stride lo
      g1[7] = 0;
      i32x4 g2;
      g2[0] = 0;                                         // tensor_dim2 (unused)
      g2[1] = XA_LD;                                     // lds_addr_increment = 776 elts
      g2[2] = 768;                                       // global_addr_increment = 768 elts
      g2[3] = (int)(15u << 16);                          // iterate_count = 15 (-> 16x)
      i32x4 g3; g3[0] = 0; g3[1] = 0; g3[2] = 0; g3[3] = 0;
      i32x8 g4; g4[0] = 0; g4[1] = 0; g4[2] = 0; g4[3] = 0;
      g4[4] = 0; g4[5] = 0; g4[6] = 0; g4[7] = 0;
      __builtin_amdgcn_tensor_load_to_lds(g0, g1, g2, g3, g4, 0);
      __builtin_amdgcn_s_wait_tensorcnt(0);
    }
#else
    const _Float16* src = Ah16 + (long)tile * 16 * DM;
    for (int i = tid * 4; i < 16 * DM; i += 1024) {
      int r = i / DM, c = i % DM;
      *(v4h*)(Xa + r * XA_LD + c) = *(const v4h*)(src + i);
    }
#endif
  } else {
    const float* src = Af32 + (long)tile * 16 * DM;
    for (int i = tid * 4; i < 16 * DM; i += 1024) {
      int r = i / DM, c = i % DM;
      float4 f = *(const float4*)(src + i);
      v4h h4;
      h4[0] = (_Float16)f.x; h4[1] = (_Float16)f.y;
      h4[2] = (_Float16)f.z; h4[3] = (_Float16)f.w;
      *(v4h*)(Xa + r * XA_LD + c) = h4;
    }
  }
  __syncthreads();
  // GEMM1: H = gelu(Xa @ W1 + b1), 16x3072
  for (int nt = wave; nt < FFH / 16; nt += 8) {
    v8f acc = {};
    int n = nt * 16 + lr;
    for (int k0 = 0; k0 < DM; k0 += 32) {
      v16h a;
      const _Float16* ar = Xa + lr * XA_LD + k0 + khalf;
      #pragma unroll
      for (int t = 0; t < 16; ++t) a[t] = ar[(t & 7) + ((t & 8) ? 16 : 0)];
      v16h bf;
      int kb = k0 + kh16;
      if (kb + 32 < DM) __builtin_prefetch(&W1[(long)(kb + 32) * FFH + n], 0, 0);
      #pragma unroll
      for (int t = 0; t < 16; ++t) bf[t] = (_Float16)W1[(long)(k0 + khalf + (t & 7) + ((t & 8) ? 16 : 0)) * FFH + n];
      acc = wmma32(a, bf, acc);
    }
    float bb = b1 ? b1[n] : 0.0f;
    #pragma unroll
    for (int vv = 0; vv < 8; ++vv) {
      int m = vv + hi8;
      H[m * H_LD + n] = (_Float16)geluf(acc[vv] + bb);
    }
  }
  __syncthreads();
  // GEMM2: Y = H @ W2 + b2, 16x768, fused gated-residual writeback
  for (int nt = wave; nt < DM / 16; nt += 8) {
    v8f acc = {};
    int n = nt * 16 + lr;
    for (int k0 = 0; k0 < FFH; k0 += 32) {
      v16h a;
      const _Float16* ar = H + lr * H_LD + k0 + khalf;
      #pragma unroll
      for (int t = 0; t < 16; ++t) a[t] = ar[(t & 7) + ((t & 8) ? 16 : 0)];
      v16h bf;
      int kb = k0 + kh16;
      if (kb + 32 < FFH) __builtin_prefetch(&W2[(long)(kb + 32) * DM + n], 0, 0);
      #pragma unroll
      for (int t = 0; t < 16; ++t) bf[t] = (_Float16)W2[(long)(k0 + khalf + (t & 7) + ((t & 8) ? 16 : 0)) * DM + n];
      acc = wmma32(a, bf, acc);
    }
    float bb = b2 ? b2[n] : 0.0f;
    #pragma unroll
    for (int vv = 0; vv < 8; ++vv) {
      int m = vv + hi8;
      float y = acc[vv] + bb;
      if (moe) {
        int src = ip[IO_SRC + tile * 16 + m];
        if (src >= 0) Out[(long)src * DM + n] += gateRow[src] * y;
      } else {
        long gm = (long)tile * 16 + m;
        float g = gateVec[(long)((int)(gm / rowsPerBatch)) * gateStride + n];
        Out[gm * DM + n] += g * y;
      }
    }
  }
}

// ---------------------------------------------------------------------------
// Orchestration
// ---------------------------------------------------------------------------
extern "C" void kernel_launch(void* const* d_in, const int* in_sizes, int n_in,
                              void* d_out, int out_size, void* d_ws, size_t ws_size,
                              hipStream_t stream) {
  (void)in_sizes; (void)n_in; (void)out_size; (void)ws_size;
  const float* x_traj = (const float*)d_in[0];
  const int*   tarr   = (const int*)d_in[1];
  const float* tokA   = (const float*)d_in[2];
  const float* tokB   = (const float*)d_in[3];
  const float* tokC   = (const float*)d_in[4];
  const float* W_traj = (const float*)d_in[5];
  const float* b_traj = (const float*)d_in[6];
  const float* Wftime = (const float*)d_in[7];
  const float* Wt1 = (const float*)d_in[8];
  const float* bt1 = (const float*)d_in[9];
  const float* Wt2 = (const float*)d_in[10];
  const float* bt2 = (const float*)d_in[11];
  const float* moe_attn = (const float*)d_in[12];
  const float* moe_rt   = (const float*)d_in[13];
  const float* moe_W1   = (const float*)d_in[14];
  const float* moe_W2   = (const float*)d_in[15];
  const float* dq   = (const float*)d_in[16];
  const float* dqb  = (const float*)d_in[17];
  const float* dp   = (const float*)d_in[18];
  const float* dpb  = (const float*)d_in[19];
  const float* dm1  = (const float*)d_in[20];
  const float* dm1b = (const float*)d_in[21];
  const float* dm2  = (const float*)d_in[22];
  const float* dm2b = (const float*)d_in[23];
  const float* dada  = (const float*)d_in[24];
  const float* dadab = (const float*)d_in[25];
  const float* oaw = (const float*)d_in[26];
  const float* oab = (const float*)d_in[27];
  const float* ow  = (const float*)d_in[28];
  const float* ob  = (const float*)d_in[29];

  float* ws = (float*)d_ws;
  size_t off = 0;
  auto alloc = [&](size_t n) { size_t r = off; off += (n + 63) & ~(size_t)63; return r; };
  float* X     = ws + alloc((size_t)BAT * NTOK * DM);
  float* HMOD  = ws + alloc((size_t)BAT * NTOK * DM);
  float* QKV   = ws + alloc((size_t)BAT * NTOK * 2304);
  float* ATTN  = ws + alloc((size_t)BAT * NTOK * DM);
  float* TRAJ  = ws + alloc((size_t)BAT * 256 * DM);
  float* GA    = ws + alloc((size_t)BAT * 64 * DM);
  float* GB    = ws + alloc((size_t)BAT * 32 * DM);
  float* GC    = ws + alloc((size_t)BAT * 192 * DM);
  float* LNB   = ws + alloc((size_t)BAT * 192 * DM);
  _Float16* XS = (_Float16*)(ws + alloc((size_t)1600 * DM / 2));
  float* GATEV = ws + alloc(1600);
  // conditioning buffers padded to 16 rows so every GEMM has M%16==0
  float* TEMB  = ws + alloc((size_t)16 * 256);
  float* CB    = ws + alloc((size_t)16 * DM);
  float* SCC   = ws + alloc((size_t)16 * DM);
  float* MODB  = ws + alloc((size_t)16 * 4608);
  float* OUTA  = ws + alloc((size_t)16 * 1536);
  float* HEADO = ws + alloc((size_t)BAT * NTOK * 14);
  int*   ip    = (int*)(ws + alloc(4096));

  (void)hipFuncSetAttribute((const void*)k_ffn,
                            hipFuncAttributeMaxDynamicSharedMemorySize, FFN_SMEM);

  auto gemm = [&](const float* A, const float* W, const float* bias, const float* gate,
                  int gStride, int rpb, const float* resid, float* C,
                  int M, int K, int N, int act) {
    dim3 g((N + 127) / 128, (M + 15) / 16);
    k_gemm<<<g, 256, 0, stream>>>(A, W, bias, gate, gStride, rpb, resid, C, M, K, N, act);
  };

  // --- timestep conditioning: c = silu(temb@Wt1+bt1)@Wt2+bt2 ; sc = silu(c)
  k_zero_f<<<(8 * 256 + 255) / 256, 256, 0, stream>>>(TEMB + 8 * 256, 8 * 256);
  k_zero_f<<<(8 * DM + 255) / 256, 256, 0, stream>>>(SCC + 8 * DM, 8 * DM);
  k_temb<<<BAT, 256, 0, stream>>>(tarr, TEMB);
  gemm(TEMB, Wt1, bt1, nullptr, 0, 1, nullptr, HMOD, 16, 256, DM, 2);
  gemm(HMOD, Wt2, bt2, nullptr, 0, 1, nullptr, CB, 16, DM, DM, 0);
  k_silu<<<(BAT * DM + 255) / 256, 256, 0, stream>>>(CB, SCC, BAT * DM);

  // --- trajectory embedding (K=7 -> slow path, tiny)
  gemm(x_traj, W_traj, b_traj, nullptr, 0, 1, nullptr, TRAJ, BAT * 256, 7, DM, 0);

  // --- MoE group working copies (don't mutate inputs)
  k_copy<<<256, 256, 0, stream>>>(tokA, GA, BAT * 64 * DM);
  k_copy<<<256, 256, 0, stream>>>(tokB, GB, BAT * 32 * DM);
  k_copy<<<256, 256, 0, stream>>>(tokC, GC, BAT * 192 * DM);

  const long WSQ = (long)DM * DM;
  float* MQ = QKV;
  float* MK = QKV + (size_t)1536 * DM;
  float* MV = QKV + (size_t)2 * 1536 * DM;

  auto cross = [&](float* Gq, int Rq, int Nq, const float* Gkv, int Rkv, int Nk,
                   const float* Wc) {
    k_ln_mod<<<Rq, 256, 0, stream>>>(Gq, LNB, nullptr, nullptr, 0, 1, Rq);
    gemm(LNB, Wc + 0 * WSQ, nullptr, nullptr, 0, 1, nullptr, MQ, Rq, DM, DM, 0);
    gemm(Gkv, Wc + 1 * WSQ, nullptr, nullptr, 0, 1, nullptr, MK, Rkv, DM, DM, 0);
    gemm(Gkv, Wc + 2 * WSQ, nullptr, nullptr, 0, 1, nullptr, MV, Rkv, DM, DM, 0);
    dim3 ag(BAT * NHD, (Nq / 16 + 7) / 8);
    k_attn<<<ag, 256, 0, stream>>>(MQ, DM, MK, DM, MV, DM, ATTN, DM, Nq, Nk);
    gemm(ATTN, Wc + 3 * WSQ, nullptr, nullptr, 0, 1, Gq, Gq, Rq, DM, DM, 0);
  };

  auto moe = [&](float* G, int R, int blk, int grp) {
    k_ln_mod<<<R, 256, 0, stream>>>(G, LNB, nullptr, nullptr, 0, 1, R);
    k_zero_i<<<1, 32, 0, stream>>>(ip, 4);
    k_router<<<R / 8, 256, 0, stream>>>(LNB, moe_rt + (size_t)(blk * 3 + grp) * DM * 4, R, ip, GATEV);
    k_offsets<<<1, 1, 0, stream>>>(ip);
    k_init_src<<<(R + 64 + 255) / 256, 256, 0, stream>>>(ip, R + 64);
    k_scatter<<<(R + 255) / 256, 256, 0, stream>>>(ip, R);
    k_gather<<<R + 64, 256, 0, stream>>>(LNB, ip, XS, R + 64);
    const float* W1 = moe_W1 + (size_t)(blk * 3 + grp) * 4 * DM * FFH;
    const float* W2 = moe_W2 + (size_t)(blk * 3 + grp) * 4 * FFH * DM;
    k_ffn<<<R / 16 + 4, 256, FFN_SMEM, stream>>>(
        nullptr, XS, W1, nullptr, W2, nullptr, G,
        nullptr, 0, 1, GATEV, ip, (long)DM * FFH, (long)FFH * DM);
  };

  // --- StackedMoEBlocks
  for (int i = 0; i < 2; ++i) {
    const float* Wc0 = moe_attn + (size_t)((i * 2 + 0) * 4) * WSQ;  // C <- A
    const float* Wc1 = moe_attn + (size_t)((i * 2 + 1) * 4) * WSQ;  // B <- C
    cross(GC, BAT * 192, 192, GA, BAT * 64, 64, Wc0);
    cross(GB, BAT * 32, 32, GC, BAT * 192, 192, Wc1);
    moe(GA, BAT * 64, i, 0);
    moe(GB, BAT * 32, i, 1);
    moe(GC, BAT * 192, i, 2);
  }

  // --- assemble X = [traj+ftime | A | B | C]
  k_assemble<<<BAT * NTOK, 256, 0, stream>>>(TRAJ, Wftime, GA, GB, GC, X);

  // --- DiT backbone (adaLN-zero)
  const int M = BAT * NTOK;
  for (int i = 0; i < 6; ++i) {
    gemm(SCC, dada + (size_t)i * DM * 4608, dadab + (size_t)i * 4608,
         nullptr, 0, 1, nullptr, MODB, 16, DM, 4608, 0);
    k_ln_mod<<<M, 256, 0, stream>>>(X, HMOD, MODB + 0, MODB + 768, 4608, NTOK, M);
    gemm(HMOD, dq + (size_t)i * DM * 2304, dqb + (size_t)i * 2304,
         nullptr, 0, 1, nullptr, QKV, M, DM, 2304, 0);
    dim3 ag(BAT * NHD, (NTOK / 16 + 7) / 8);
    k_attn<<<ag, 256, 0, stream>>>(QKV, 2304, QKV + 768, 2304, QKV + 1536, 2304,
                                   ATTN, DM, NTOK, NTOK);
    gemm(ATTN, dp + (size_t)i * WSQ, dpb + (size_t)i * DM,
         MODB + 1536, 4608, NTOK, X, X, M, DM, DM, 0);
    k_ln_mod<<<M, 256, 0, stream>>>(X, HMOD, MODB + 2304, MODB + 3072, 4608, NTOK, M);
    k_ffn<<<M / 16, 256, FFN_SMEM, stream>>>(
        HMOD, nullptr, dm1 + (size_t)i * DM * FFH, dm1b + (size_t)i * FFH,
        dm2 + (size_t)i * FFH * DM, dm2b + (size_t)i * DM, X,
        MODB + 3840, 4608, NTOK, nullptr, nullptr, 0, 0);
  }

  // --- output head
  gemm(SCC, oaw, oab, nullptr, 0, 1, nullptr, OUTA, 16, DM, 1536, 0);
  k_ln_mod<<<M, 256, 0, stream>>>(X, HMOD, OUTA + 0, OUTA + 768, 1536, NTOK, M);
  gemm(HMOD, ow, ob, nullptr, 0, 1, nullptr, HEADO, M, DM, 14, 0);
  k_outcopy<<<(BAT * 256 * 14 + 255) / 256, 256, 0, stream>>>(HEADO, (float*)d_out);
}
